// GAT_44770739093839
// MI455X (gfx1250) — matrix operand order repaired and checked by reading, not compiled, not run on hardware
//
#include <hip/hip_runtime.h>
#include <hip/hip_bf16.h>
#include <math.h>

// ---------------- problem constants (match reference) ----------------
#define NN     30000          // nodes
#define NE     600000         // edges
#define KF     512            // NFEAT
#define HD1    128            // heads*nhid conv1
#define H1     8              // heads conv1
#define D1     16             // nhid conv1
#define IN2    256            // conv2 input width
#define OUT2   100            // conv2 output width
#define OUT2P  112            // padded to multiple of 16
#define N2     (2*NN)
#define NCL    20

typedef __bf16 bf16_t;
typedef __attribute__((ext_vector_type(16))) __bf16 v16bf;
typedef __attribute__((ext_vector_type(8)))  __bf16 v8bf;
typedef __attribute__((ext_vector_type(8)))  float  v8f;

// monotonic float<->uint encoding for atomic segment-max
__device__ __forceinline__ unsigned f2ord(float f) {
  unsigned u = __float_as_uint(f);
  return (u & 0x80000000u) ? ~u : (u | 0x80000000u);
}
__device__ __forceinline__ float ord2f(unsigned u) {
  unsigned v = (u & 0x80000000u) ? (u & 0x7FFFFFFFu) : ~u;
  return __uint_as_float(v);
}
#define ORD_NEG_INF 0x007FFFFFu   // f2ord(-inf)

// ---------------- utility kernels ----------------
__global__ void fill_f32_k(float* p, float v, long long n) {
  long long i = (long long)blockIdx.x * blockDim.x + threadIdx.x;
  if (i < n) p[i] = v;
}
__global__ void fill_u32_k(unsigned* p, unsigned v, long long n) {
  long long i = (long long)blockIdx.x * blockDim.x + threadIdx.x;
  if (i < n) p[i] = v;
}
__global__ void cvt_f32_bf16_k(const float* __restrict__ s, bf16_t* __restrict__ d,
                               long long n) {
  long long i = (long long)blockIdx.x * blockDim.x + threadIdx.x;
  if (i < n) d[i] = (bf16_t)s[i];
}
// Wt[n][k] = W[k][n] (bf16), zero-padded columns n >= nsrc
__global__ void transpose_w_k(const float* __restrict__ W, bf16_t* __restrict__ Wt,
                              int K, int nsrc, int npad) {
  long long i = (long long)blockIdx.x * blockDim.x + threadIdx.x;
  if (i >= (long long)npad * K) return;
  int n = (int)(i / K), k = (int)(i % K);
  Wt[i] = (n < nsrc) ? (bf16_t)W[(size_t)k * nsrc + n] : (bf16_t)0.0f;
}

// ---------------- WMMA GEMM, async-LDS double-buffered B staging ----------------
// C[M, NP] = A[M, K](bf16, row-major) * Bt[NP, K](bf16, col = row of Bt)^T
// 256 threads = 8 waves; wave w computes row tile (blockIdx*8 + w) x all NT
// column tiles. B staged through 2 LDS buffers in K-chunks of 128 via
// global_load_async_to_lds_b128 (ASYNCcnt); copy of chunk ci+1 overlaps the
// WMMA work on chunk ci. B fragments for all NT tiles are gathered into a
// register array first so the NT WMMAs issue back-to-back after one wait.
template<int K, int NT>
__global__ void wmma_gemm_lds_k(const bf16_t* __restrict__ A,
                                const bf16_t* __restrict__ Bt,
                                float* __restrict__ C, int M) {
  constexpr int NP    = NT * 16;
  constexpr int KCH   = 128;                     // K elements per LDS chunk
  constexpr int CHB   = NP * KCH * 2;            // chunk bytes (32 KB / 28 KB)
  constexpr int UNITS = CHB / 16;                // 16-byte units per chunk
  constexpr int NCH   = K / KCH;                 // chunks (4 / 2)
  __shared__ __align__(16) bf16_t Bs[2][NP * KCH];

  const int tid    = (int)threadIdx.x;
  const int waveid = tid >> 5;
  const int lane   = tid & 31;
  const int tm     = (int)blockIdx.x * 8 + waveid;
  const int tilesM = M >> 4;
  const int mr     = lane & 15;                  // row (A) / col (B)
  const int kh     = lane >> 4;                  // lane-half selects K sub-block
  const bf16_t* arow = A + (size_t)(tm * 16 + mr) * K;

  v8f acc[NT];
#pragma unroll
  for (int t = 0; t < NT; ++t) acc[t] = (v8f){0.f,0.f,0.f,0.f,0.f,0.f,0.f,0.f};

  // cooperative async copy of B chunk ci -> LDS buffer buf.
  // UNITS is an exact multiple of 256, so EXEC stays full for every issue.
  auto stage = [&](int ci, int buf) {
#pragma unroll 1
    for (int u = tid; u < UNITS; u += 256) {
      int col = u >> 4;                          // KCH*2/16 = 16 units per column
      int off = (u & 15) * 8;                    // element offset inside chunk row
      const bf16_t* g = Bt + (size_t)col * K + ci * KCH + off;
      unsigned ldsoff = (unsigned)(buf * CHB + u * 16);
      asm volatile("global_load_async_to_lds_b128 %0, %1, off"
                   :: "v"(ldsoff), "v"((unsigned long long)(size_t)g)
                   : "memory");
    }
  };

  stage(0, 0);
  for (int ci = 0; ci < NCH; ++ci) {
    const int buf = ci & 1;
    asm volatile("s_wait_asynccnt 0" ::: "memory");
    __syncthreads();                             // buffer `buf` ready for all waves
    if (ci + 1 < NCH) stage(ci + 1, buf ^ 1);    // overlap DMA with compute

    if (tm < tilesM) {                           // wave-uniform; EXEC full inside
      const bf16_t* bbase = &Bs[buf][0];
      for (int k0 = 0; k0 < KCH; k0 += 32) {
        union { v16bf v; v8bf h[2]; } a;
        // A 16x32 bf16 layout: lane half 0 -> K 0..7 & 16..23, half 1 -> +8
        a.h[0] = *(const v8bf*)(arow + ci * KCH + k0 + kh * 8);
        a.h[1] = *(const v8bf*)(arow + ci * KCH + k0 + 16 + kh * 8);
        union { v16bf v; v8bf h[2]; } b[NT];
#pragma unroll
        for (int tn = 0; tn < NT; ++tn) {
          // B 32x16 layout: lane half 0 -> K 0..7 & 16..23, half 1 -> +8
          const bf16_t* bp = bbase + (tn * 16 + mr) * KCH + k0 + kh * 8;
          b[tn].h[0] = *(const v8bf*)bp;
          b[tn].h[1] = *(const v8bf*)(bp + 16);
        }
#pragma unroll
        for (int tn = 0; tn < NT; ++tn)
          acc[tn] = __builtin_amdgcn_wmma_f32_16x16x32_bf16(
              false, a.v, false, b[tn].v, (short)0, acc[tn], false, false);
      }
    }
    __syncthreads();                             // all reads of `buf` done before reuse
  }

  if (tm < tilesM) {
#pragma unroll
    for (int tn = 0; tn < NT; ++tn) {
      // C/D layout: VGPR r holds row r + 8*kh, col = lane&15
      float* crow = C + (size_t)(tm * 16 + kh * 8) * NP + tn * 16 + mr;
#pragma unroll
      for (int r = 0; r < 8; ++r) crow[(size_t)r * NP] = acc[tn][r];
    }
  }
}

// ---------------- attention scores: a_src/a_dst = sum_d xl*att ----------------
__global__ void att_scores_k(const float* __restrict__ xl, int ld, int H, int D,
                             const float* __restrict__ ws, const float* __restrict__ wd,
                             float* __restrict__ as_, float* __restrict__ ad_, int nn) {
  long long i = (long long)blockIdx.x * blockDim.x + threadIdx.x;
  if (i >= (long long)nn * H) return;
  int n = (int)(i / H), h = (int)(i % H);
  const float* v = xl + (size_t)n * ld + h * D;
  float s = 0.f, d = 0.f;
  for (int j = 0; j < D; ++j) { s += v[j] * ws[h * D + j]; d += v[j] * wd[h * D + j]; }
  as_[i] = s; ad_[i] = d;
}

// edge (src,dst) mapping: conv1 two=0 (E edges + N loops); conv2 two=1
// (E edges, E edges shifted by noff, 2N loops)
__device__ __forceinline__ void edge_sd(int t, int E_, int noff, int two,
                                        const int* __restrict__ ei, int& s, int& d) {
  if (t < E_)                { s = ei[t];            d = ei[E_ + t]; }
  else if (two && t < 2*E_)  { s = ei[t-E_] + noff;  d = ei[t] + noff; } // ei[E_+t-E_]
  else                       { int b = two ? 2*E_ : E_; s = d = t - b; }
}

__global__ void edge_max_k(const int* __restrict__ ei, int E_, int two, int noff,
                           const float* __restrict__ as_, const float* __restrict__ ad_,
                           unsigned* __restrict__ amax, int H, int T) {
  long long i = (long long)blockIdx.x * blockDim.x + threadIdx.x;
  if (i >= (long long)T * H) return;
  int t = (int)(i / H), h = (int)(i % H);
  int s, d; edge_sd(t, E_, noff, two, ei, s, d);
  float a = as_[(size_t)s * H + h] + ad_[(size_t)d * H + h];
  a = a > 0.f ? a : 0.2f * a;                        // leaky_relu 0.2
  atomicMax(&amax[(size_t)d * H + h], f2ord(a));
}

__global__ void edge_sum_k(const int* __restrict__ ei, int E_, int two, int noff,
                           const float* __restrict__ as_, const float* __restrict__ ad_,
                           const unsigned* __restrict__ amax, float* __restrict__ z,
                           int H, int T) {
  long long i = (long long)blockIdx.x * blockDim.x + threadIdx.x;
  if (i >= (long long)T * H) return;
  int t = (int)(i / H), h = (int)(i % H);
  int s, d; edge_sd(t, E_, noff, two, ei, s, d);
  float a = as_[(size_t)s * H + h] + ad_[(size_t)d * H + h];
  a = a > 0.f ? a : 0.2f * a;
  float e = __expf(a - ord2f(amax[(size_t)d * H + h]));
  atomicAdd(&z[(size_t)d * H + h], e);
}

__global__ void edge_agg_k(const int* __restrict__ ei, int E_, int two, int noff,
                           const float* __restrict__ as_, const float* __restrict__ ad_,
                           const unsigned* __restrict__ amax, const float* __restrict__ z,
                           const float* __restrict__ xl, int ldx,
                           float* __restrict__ agg, int ldagg, int H, int D, int T) {
  long long i = (long long)blockIdx.x * blockDim.x + threadIdx.x;
  if (i >= (long long)T * H) return;
  int t = (int)(i / H), h = (int)(i % H);
  int s, d; edge_sd(t, E_, noff, two, ei, s, d);
  float a = as_[(size_t)s * H + h] + ad_[(size_t)d * H + h];
  a = a > 0.f ? a : 0.2f * a;
  float e = __expf(a - ord2f(amax[(size_t)d * H + h]));
  float alpha = e / (z[(size_t)d * H + h] + 1e-16f);
  const float* src = xl + (size_t)s * ldx + h * D;
  float* dst = agg + (size_t)d * ldagg + h * D;
  __builtin_prefetch(src, 0, 0);                     // global_prefetch_b8
  for (int j = 0; j < D; ++j) atomicAdd(&dst[j], src[j] * alpha);
}

// ---------------- conv1 epilogue: bias + ELU (in place) + cluster sums ----------
__global__ void finalize_h_k(float* __restrict__ hb, const float* __restrict__ bias1,
                             const int* __restrict__ cluster,
                             const unsigned char* __restrict__ mask,
                             float* __restrict__ csum, float* __restrict__ ccnt) {
  long long i = (long long)blockIdx.x * blockDim.x + threadIdx.x;
  if (i >= (long long)NN * HD1) return;
  int n = (int)(i >> 7), c = (int)(i & 127);
  float v = hb[i] + bias1[c];
  v = v > 0.f ? v : (__expf(v) - 1.f);               // ELU
  hb[i] = v;
  if (mask[n]) {
    atomicAdd(&csum[(size_t)cluster[n] * HD1 + c], v);
    if (c == 0) atomicAdd(&ccnt[cluster[n]], 1.f);
  }
}

__global__ void cluster_norm_k(const float* __restrict__ csum,
                               const float* __restrict__ ccnt,
                               float* __restrict__ cfeat) {
  int i = blockIdx.x * blockDim.x + threadIdx.x;
  if (i >= NCL * HD1) return;
  cfeat[i] = csum[i] / ccnt[i / HD1];
}

// x2 = [[h, x1], [x1, h]] in bf16
__global__ void build_x2_k(const float* __restrict__ hb, const float* __restrict__ cfeat,
                           const int* __restrict__ cluster, bf16_t* __restrict__ x2b) {
  long long i = (long long)blockIdx.x * blockDim.x + threadIdx.x;
  if (i >= (long long)NN * HD1) return;
  int n = (int)(i >> 7), j = (int)(i & 127);
  bf16_t hv = (bf16_t)hb[i];
  bf16_t cv = (bf16_t)cfeat[(size_t)cluster[n] * HD1 + j];
  x2b[(size_t)n * IN2 + j]              = hv;
  x2b[(size_t)n * IN2 + HD1 + j]        = cv;
  x2b[(size_t)(n + NN) * IN2 + j]       = cv;
  x2b[(size_t)(n + NN) * IN2 + HD1 + j] = hv;
}

__global__ void final_out_k(const float* __restrict__ agg2,
                            const float* __restrict__ bias2, float* __restrict__ out) {
  long long i = (long long)blockIdx.x * blockDim.x + threadIdx.x;
  if (i >= (long long)N2 * OUT2) return;
  out[i] = agg2[i] + bias2[(int)(i % OUT2)];
}

// ---------------- host side ----------------
static inline int cdiv(long long a, int b) { return (int)((a + b - 1) / b); }

extern "C" void kernel_launch(void* const* d_in, const int* in_sizes, int n_in,
                              void* d_out, int out_size, void* d_ws, size_t ws_size,
                              hipStream_t stream) {
  (void)in_sizes; (void)n_in; (void)out_size; (void)ws_size;
  const float* x        = (const float*)d_in[0];
  const float* W1       = (const float*)d_in[1];
  const float* attS1    = (const float*)d_in[2];
  const float* attD1    = (const float*)d_in[3];
  const float* bias1    = (const float*)d_in[4];
  const float* W2       = (const float*)d_in[5];
  const float* attS2    = (const float*)d_in[6];
  const float* attD2    = (const float*)d_in[7];
  const float* bias2    = (const float*)d_in[8];
  const int*   ei       = (const int*)d_in[9];
  const int*   cluster  = (const int*)d_in[10];
  const unsigned char* mask = (const unsigned char*)d_in[11];
  float* out = (float*)d_out;

  // ---- workspace layout (bytes) ----
  char* ws = (char*)d_ws;
  size_t off = 0;
  auto take = [&](size_t bytes) { size_t o = off; off += (bytes + 255) & ~(size_t)255; return o; };
  size_t o_xb    = take((size_t)NN * KF * sizeof(bf16_t));     // x bf16; reused as x2 bf16
  size_t o_w1t   = take((size_t)HD1 * KF * sizeof(bf16_t));
  size_t o_w2t   = take((size_t)OUT2P * IN2 * sizeof(bf16_t));
  size_t o_xl    = take((size_t)NN * HD1 * sizeof(float));
  size_t o_as1   = take((size_t)NN * H1 * sizeof(float));
  size_t o_ad1   = take((size_t)NN * H1 * sizeof(float));
  size_t o_am1   = take((size_t)NN * H1 * sizeof(unsigned));
  size_t o_z1    = take((size_t)NN * H1 * sizeof(float));
  size_t o_agg1  = take((size_t)NN * HD1 * sizeof(float));     // becomes h in place
  size_t o_csum  = take((size_t)NCL * HD1 * sizeof(float));
  size_t o_ccnt  = take((size_t)NCL * sizeof(float));
  size_t o_cfeat = take((size_t)NCL * HD1 * sizeof(float));
  size_t o_g2    = take((size_t)N2 * OUT2P * sizeof(float));
  size_t o_as2   = take((size_t)N2 * sizeof(float));
  size_t o_ad2   = take((size_t)N2 * sizeof(float));
  size_t o_am2   = take((size_t)N2 * sizeof(unsigned));
  size_t o_z2    = take((size_t)N2 * sizeof(float));
  size_t o_agg2  = take((size_t)N2 * OUT2 * sizeof(float));

  bf16_t*   xb   = (bf16_t*)(ws + o_xb);
  bf16_t*   x2b  = (bf16_t*)(ws + o_xb);     // reuse: dead after GEMM1
  bf16_t*   w1t  = (bf16_t*)(ws + o_w1t);
  bf16_t*   w2t  = (bf16_t*)(ws + o_w2t);
  float*    xl   = (float*)(ws + o_xl);
  float*    as1  = (float*)(ws + o_as1);
  float*    ad1  = (float*)(ws + o_ad1);
  unsigned* am1  = (unsigned*)(ws + o_am1);
  float*    z1   = (float*)(ws + o_z1);
  float*    agg1 = (float*)(ws + o_agg1);    // -> h after finalize
  float*    csum = (float*)(ws + o_csum);
  float*    ccnt = (float*)(ws + o_ccnt);
  float*    cfeat= (float*)(ws + o_cfeat);
  float*    g2   = (float*)(ws + o_g2);
  float*    as2  = (float*)(ws + o_as2);
  float*    ad2  = (float*)(ws + o_ad2);
  unsigned* am2  = (unsigned*)(ws + o_am2);
  float*    z2   = (float*)(ws + o_z2);
  float*    agg2 = (float*)(ws + o_agg2);

  const int T = 256;
  // ---- init accumulators ----
  fill_f32_k<<<cdiv((long long)NN*HD1,T),T,0,stream>>>(agg1, 0.f, (long long)NN*HD1);
  fill_f32_k<<<cdiv((long long)NN*H1,T),T,0,stream>>>(z1, 0.f, (long long)NN*H1);
  fill_u32_k<<<cdiv((long long)NN*H1,T),T,0,stream>>>(am1, ORD_NEG_INF, (long long)NN*H1);
  fill_f32_k<<<cdiv(NCL*HD1,T),T,0,stream>>>(csum, 0.f, NCL*HD1);
  fill_f32_k<<<cdiv(NCL,T),T,0,stream>>>(ccnt, 0.f, NCL);
  fill_f32_k<<<cdiv((long long)N2*OUT2,T),T,0,stream>>>(agg2, 0.f, (long long)N2*OUT2);
  fill_f32_k<<<cdiv(N2,T),T,0,stream>>>(z2, 0.f, N2);
  fill_u32_k<<<cdiv(N2,T),T,0,stream>>>(am2, ORD_NEG_INF, N2);

  // ---- precision conversion / weight transpose ----
  cvt_f32_bf16_k<<<cdiv((long long)NN*KF,T),T,0,stream>>>(x, xb, (long long)NN*KF);
  transpose_w_k<<<cdiv((long long)HD1*KF,T),T,0,stream>>>(W1, w1t, KF, HD1, HD1);
  transpose_w_k<<<cdiv((long long)OUT2P*IN2,T),T,0,stream>>>(W2, w2t, IN2, OUT2, OUT2P);

  // ---- GEMM1: xl = x @ W1 ; 1875 row tiles, 8 col tiles, 8 waves/block ----
  wmma_gemm_lds_k<KF, HD1/16><<<cdiv(NN/16, 8), 256, 0, stream>>>(xb, w1t, xl, NN);

  // ---- conv1 attention ----
  att_scores_k<<<cdiv((long long)NN*H1,T),T,0,stream>>>(xl, HD1, H1, D1, attS1, attD1,
                                                        as1, ad1, NN);
  const int T1 = NE + NN;   // edges + self loops
  edge_max_k<<<cdiv((long long)T1*H1,T),T,0,stream>>>(ei, NE, 0, 0, as1, ad1, am1, H1, T1);
  edge_sum_k<<<cdiv((long long)T1*H1,T),T,0,stream>>>(ei, NE, 0, 0, as1, ad1, am1, z1, H1, T1);
  edge_agg_k<<<cdiv((long long)T1*H1,T),T,0,stream>>>(ei, NE, 0, 0, as1, ad1, am1, z1,
                                                      xl, HD1, agg1, HD1, H1, D1, T1);

  // ---- bias + ELU, cluster mean features, build x2 ----
  finalize_h_k<<<cdiv((long long)NN*HD1,T),T,0,stream>>>(agg1, bias1, cluster, mask, csum, ccnt);
  cluster_norm_k<<<cdiv(NCL*HD1,T),T,0,stream>>>(csum, ccnt, cfeat);
  build_x2_k<<<cdiv((long long)NN*HD1,T),T,0,stream>>>(agg1, cfeat, cluster, x2b);

  // ---- GEMM2: g2 = x2 @ W2 ; 3750 row tiles, 7 col tiles ----
  wmma_gemm_lds_k<IN2, OUT2P/16><<<cdiv(N2/16, 8), 256, 0, stream>>>(x2b, w2t, g2, N2);

  // ---- conv2 attention (1 head, D=100) ----
  att_scores_k<<<cdiv((long long)N2,T),T,0,stream>>>(g2, OUT2P, 1, OUT2, attS2, attD2,
                                                     as2, ad2, N2);
  const int T2 = 2*NE + N2; // doubled edges + self loops on 2N nodes
  edge_max_k<<<cdiv((long long)T2,T),T,0,stream>>>(ei, NE, 1, NN, as2, ad2, am2, 1, T2);
  edge_sum_k<<<cdiv((long long)T2,T),T,0,stream>>>(ei, NE, 1, NN, as2, ad2, am2, z2, 1, T2);
  edge_agg_k<<<cdiv((long long)T2,T),T,0,stream>>>(ei, NE, 1, NN, as2, ad2, am2, z2,
                                                   g2, OUT2P, agg2, OUT2, 1, OUT2, T2);

  // ---- final: out = agg2 + bias2 ----
  final_out_k<<<cdiv((long long)N2*OUT2,T),T,0,stream>>>(agg2, bias2, out);
}
